// SiameseGraphNetworkGCN_v2_47321949667572
// MI455X (gfx1250) — matrix operand order repaired and checked by reading, not compiled, not run on hardware
//
#include <hip/hip_runtime.h>
#include <hip/hip_bf16.h>
#include <math.h>

#define NN 50000
#define NE 800000
#define NT (NE + NN)   // edges + self loops
#define NG 64
#define NH 5
#define NCLS 100
#define EPSV 1e-5f

typedef __attribute__((ext_vector_type(16))) __bf16 v16bf;
typedef __attribute__((ext_vector_type(8)))  __bf16 v8bf;
typedef __attribute__((ext_vector_type(8)))  float  v8f;

// ---------------- device helpers ----------------
__device__ __forceinline__ float gelu_exact(float v) {
  return 0.5f * v * (1.0f + erff(v * 0.70710678118654752f));
}
__device__ __forceinline__ unsigned flip_f(float f) {
  unsigned u = __float_as_uint(f);
  return (u & 0x80000000u) ? ~u : (u | 0x80000000u);
}
__device__ __forceinline__ float unflip_f(unsigned u) {
  u = (u & 0x80000000u) ? (u ^ 0x80000000u) : ~u;
  return __uint_as_float(u);
}
__device__ __forceinline__ void edge_sd(long long e, const long long* ei, int& s, int& d) {
  if (e < NE) { s = (int)ei[e]; d = (int)ei[NE + e]; }
  else        { s = d = (int)(e - NE); }
}

// ---------------- generic fills / converts ----------------
__global__ void k_fill_f32(float* p, float v, long long n) {
  for (long long i = blockIdx.x * (long long)blockDim.x + threadIdx.x; i < n;
       i += (long long)gridDim.x * blockDim.x) p[i] = v;
}
__global__ void k_fill_u32(unsigned* p, unsigned v, long long n) {
  for (long long i = blockIdx.x * (long long)blockDim.x + threadIdx.x; i < n;
       i += (long long)gridDim.x * blockDim.x) p[i] = v;
}
__global__ void k_to_bf16(__bf16* dst, const float* src, long long n) {
  for (long long i = blockIdx.x * (long long)blockDim.x + threadIdx.x; i < n;
       i += (long long)gridDim.x * blockDim.x) dst[i] = (__bf16)src[i];
}

// Pack B [128 x N] f32 (leading dim src_ld, column offset col0) into bf16 WMMA
// fragment order so GEMM B loads are contiguous per lane:
//   dst[(((kb*tilesN)+tn)*32 + lane)*16 + i] =
//       B[(kb*32 + (lane>>4)*16 + i), col0 + tn*16 + (lane&15)]
__global__ void k_pack_b_wmma(__bf16* dst, const float* src, int N, int src_ld, int col0) {
  long long n = (long long)128 * N;
  int tilesN = N >> 4;
  for (long long i = blockIdx.x * (long long)blockDim.x + threadIdx.x; i < n;
       i += (long long)gridDim.x * blockDim.x) {
    int i16  = (int)(i & 15);
    int lane = (int)(i >> 4) & 31;
    int t    = (int)(i >> 9);          // kb*tilesN + tn
    int kb = t / tilesN, tn = t % tilesN;
    int row = kb * 32 + (lane >> 4) * 16 + i16;
    int col = tn * 16 + (lane & 15);
    dst[i] = (__bf16)src[(size_t)row * src_ld + col0 + col];
  }
}

// ---------------- degree / normalization ----------------
__global__ void k_deg_count(const long long* ei, float* deg) {
  for (long long e = blockIdx.x * (long long)blockDim.x + threadIdx.x; e < NE;
       e += (long long)gridDim.x * blockDim.x)
    atomicAdd(&deg[(int)ei[NE + e]], 1.0f);
}
__global__ void k_rsqrt_inplace(float* p, long long n) {
  for (long long i = blockIdx.x * (long long)blockDim.x + threadIdx.x; i < n;
       i += (long long)gridDim.x * blockDim.x) p[i] = rsqrtf(p[i]);
}

// ---------------- edge aggregation ----------------
__global__ void k_agg3(const long long* ei, const float* dis, const float* x, float* out) {
  for (long long e = blockIdx.x * (long long)blockDim.x + threadIdx.x; e < NT;
       e += (long long)gridDim.x * blockDim.x) {
    int s, d; edge_sd(e, ei, s, d);
    float w = dis[s] * dis[d];
    atomicAdd(&out[(size_t)d * 3 + 0], w * x[(size_t)s * 3 + 0]);
    atomicAdd(&out[(size_t)d * 3 + 1], w * x[(size_t)s * 3 + 1]);
    atomicAdd(&out[(size_t)d * 3 + 2], w * x[(size_t)s * 3 + 2]);
  }
}
// 128-wide GCN aggregation: 2 edges per 256-thread block
__global__ void k_agg128(const long long* ei, const float* dis, const float* feat, float* out) {
  long long e = (long long)blockIdx.x * 2 + (threadIdx.x >> 7);
  if (e >= NT) return;
  int c = threadIdx.x & 127;
  int s, d; edge_sd(e, ei, s, d);
  float w = dis[s] * dis[d];
  atomicAdd(&out[(size_t)d * 128 + c], w * feat[(size_t)s * 128 + c]);
}
// 128-wide GAT aggregation for one head
__global__ void k_agg_gat(const long long* ei, const float* coef, const float* denom,
                          const float* feat, float* out, int hd) {
  long long e = (long long)blockIdx.x * 2 + (threadIdx.x >> 7);
  if (e >= NT) return;
  int c = threadIdx.x & 127;
  int s, d; edge_sd(e, ei, s, d);
  float sc = coef[e * NH + hd] / fmaxf(denom[(size_t)d * NH + hd], 1e-16f);
  atomicAdd(&out[(size_t)d * 128 + c], sc * feat[(size_t)s * 128 + c]);
}

// ---------------- small input GEMM (K=3) ----------------
__global__ void k_lin3(const float* g3, const float* w_in, float* h) {
  long long n = (long long)NN * 128;
  for (long long i = blockIdx.x * (long long)blockDim.x + threadIdx.x; i < n;
       i += (long long)gridDim.x * blockDim.x) {
    int node = (int)(i >> 7), c = (int)(i & 127);
    const float* g = g3 + (size_t)node * 3;
    h[i] = g[0] * w_in[c] + g[1] * w_in[128 + c] + g[2] * w_in[256 + c];
  }
}

// ---------------- WMMA bf16 GEMM, K = 128 fixed ----------------
// C = scale*(A@B) [+ C_old].  A: [M,128] bf16 row-major.  Bp: fragment-packed.
// One 16x16 tile per wave, 8 waves per block, K fully unrolled -> 4 WMMAs.
__global__ void k_gemm128_bf16_wmma(const __bf16* __restrict__ A,
                                    const __bf16* __restrict__ Bp,
                                    float* __restrict__ C, int M, int N,
                                    float scale, int accum) {
  int tilesN = N >> 4;
  int totalTiles = (M >> 4) * tilesN;
  int tile = blockIdx.x * (blockDim.x >> 5) + (threadIdx.x >> 5);
  if (tile >= totalTiles) return;
  int tm = tile / tilesN, tn = tile % tilesN;
  int lane = threadIdx.x & 31;
  int r    = lane & 15;        // A row-in-tile ; C col-in-tile
  int hi   = lane >> 4;

  const __bf16* Arow = A + (size_t)(tm * 16 + r) * 128;
  int rbase = hi * 8;          // C rows 0..7 or 8..15
  float* Cp = C + (size_t)(tm * 16 + rbase) * N + tn * 16 + r;

  v8f acc = {0.f, 0.f, 0.f, 0.f, 0.f, 0.f, 0.f, 0.f};
  if (accum) {
    float inv = 1.0f / scale;
#pragma unroll
    for (int j = 0; j < 8; ++j) acc[j] = Cp[(size_t)j * N] * inv;
  }
#pragma unroll
  for (int kb = 0; kb < 4; ++kb) {
    // A fragment: lanes 0-15 hold K 0..15 ; lanes 16-31 hold K 16..31
    v8bf alo = *(const v8bf*)(Arow + kb * 32 + hi * 16);       // 16B aligned b128
    v8bf ahi = *(const v8bf*)(Arow + kb * 32 + hi * 16 + 8);   // 16B aligned b128
    v16bf a = __builtin_shufflevector(alo, ahi, 0, 1, 2, 3, 4, 5, 6, 7,
                                      8, 9, 10, 11, 12, 13, 14, 15);
    // B fragment: pre-packed contiguous 32B per lane (2x b128)
    v16bf b = *(const v16bf*)(Bp + ((((size_t)kb * tilesN + tn) * 32) + lane) * 16);
    acc = __builtin_amdgcn_wmma_f32_16x16x32_bf16(false, a, false, b, (short)0, acc,
                                                  false, false);
  }
#pragma unroll
  for (int j = 0; j < 8; ++j) Cp[(size_t)j * N] = acc[j] * scale;
}

// ---------------- batch norm + gelu (+residual) ----------------
__global__ void k_bn_stats(const float* x, int C, float* mu, float* var) {
  int col = blockIdx.x;
  __shared__ float ss[256], sq[256];
  float s = 0.f, q = 0.f;
  for (int r = threadIdx.x; r < NN; r += 256) {
    float v = x[(size_t)r * C + col]; s += v; q += v * v;
  }
  ss[threadIdx.x] = s; sq[threadIdx.x] = q; __syncthreads();
  for (int o = 128; o > 0; o >>= 1) {
    if ((int)threadIdx.x < o) { ss[threadIdx.x] += ss[threadIdx.x + o]; sq[threadIdx.x] += sq[threadIdx.x + o]; }
    __syncthreads();
  }
  if (threadIdx.x == 0) {
    float m = ss[0] / (float)NN;
    mu[col] = m; var[col] = sq[0] / (float)NN - m * m;
  }
}
__global__ void k_bn_gelu(const float* x, float* out, const float* res,
                          const float* mu, const float* var,
                          const float* gam, const float* bet, int C, long long n) {
  for (long long i = blockIdx.x * (long long)blockDim.x + threadIdx.x; i < n;
       i += (long long)gridDim.x * blockDim.x) {
    int c = (int)(i % C);
    float v = (x[i] - mu[c]) * rsqrtf(var[c] + EPSV) * gam[c] + bet[c];
    v = gelu_exact(v);
    out[i] = res ? v + res[i] : v;
  }
}

// ---------------- GAT attention ----------------
__global__ void k_wa(const float* wg, const float* as_, const float* ad_,
                     float* was, float* wad) {
  int i = blockIdx.x * blockDim.x + threadIdx.x;
  if (i >= 128 * NH) return;
  int k = i / NH, hd = i % NH;
  float s = 0.f, d = 0.f;
  for (int c = 0; c < 128; ++c) {
    float w = wg[(size_t)k * (NH * 128) + hd * 128 + c];
    s += w * as_[hd * 128 + c];
    d += w * ad_[hd * 128 + c];
  }
  was[i] = s; wad[i] = d;
}
__global__ void k_ah(const float* h, const float* was, const float* wad,
                     float* asrc, float* adst) {
  long long n = (long long)NN * NH;
  for (long long i = blockIdx.x * (long long)blockDim.x + threadIdx.x; i < n;
       i += (long long)gridDim.x * blockDim.x) {
    int node = (int)(i / NH), hd = (int)(i % NH);
    float s = 0.f, d = 0.f;
    const float* hr = h + (size_t)node * 128;
    for (int k = 0; k < 128; ++k) {
      float v = hr[k];
      s += v * was[k * NH + hd];
      d += v * wad[k * NH + hd];
    }
    asrc[i] = s; adst[i] = d;
  }
}
__global__ void k_att_lrelu_max(const long long* ei, const float* asrc, const float* adst,
                                float* coef, unsigned* mflip) {
  long long n = (long long)NT * NH;
  for (long long i = blockIdx.x * (long long)blockDim.x + threadIdx.x; i < n;
       i += (long long)gridDim.x * blockDim.x) {
    long long e = i / NH; int hd = (int)(i % NH);
    int s, d; edge_sd(e, ei, s, d);
    float v = asrc[(size_t)s * NH + hd] + adst[(size_t)d * NH + hd];
    v = v > 0.f ? v : 0.2f * v;
    coef[i] = v;
    atomicMax(&mflip[(size_t)d * NH + hd], flip_f(v));
  }
}
__global__ void k_att_alpha(const long long* ei, float* coef, const unsigned* mflip,
                            float* denom) {
  long long n = (long long)NT * NH;
  for (long long i = blockIdx.x * (long long)blockDim.x + threadIdx.x; i < n;
       i += (long long)gridDim.x * blockDim.x) {
    long long e = i / NH; int hd = (int)(i % NH);
    int s, d; edge_sd(e, ei, s, d); (void)s;
    float a = expf(coef[i] - unflip_f(mflip[(size_t)d * NH + hd]));
    coef[i] = a;
    atomicAdd(&denom[(size_t)d * NH + hd], a);
  }
}
__global__ void k_bias_add128(const float* src, const float* b, float* dst) {
  long long n = (long long)NN * 128;
  for (long long i = blockIdx.x * (long long)blockDim.x + threadIdx.x; i < n;
       i += (long long)gridDim.x * blockDim.x)
    dst[i] = src[i] + b[i & 127];
}

// ---------------- graph layer norm (global mean/var) ----------------
__global__ void k_red2(const float* x, long long n, float* red) {
  __shared__ float ss[256], sq[256];
  float s = 0.f, q = 0.f;
  for (long long i = blockIdx.x * (long long)blockDim.x + threadIdx.x; i < n;
       i += (long long)gridDim.x * blockDim.x) { float v = x[i]; s += v; q += v * v; }
  ss[threadIdx.x] = s; sq[threadIdx.x] = q; __syncthreads();
  for (int o = 128; o > 0; o >>= 1) {
    if ((int)threadIdx.x < o) { ss[threadIdx.x] += ss[threadIdx.x + o]; sq[threadIdx.x] += sq[threadIdx.x + o]; }
    __syncthreads();
  }
  if (threadIdx.x == 0) { atomicAdd(&red[0], ss[0]); atomicAdd(&red[1], sq[0]); }
}
__global__ void k_gln_apply(float* x, const float* red, const float* gam, const float* bet) {
  const float nTot = (float)NN * 512.0f;
  long long n = (long long)NN * 512;
  for (long long i = blockIdx.x * (long long)blockDim.x + threadIdx.x; i < n;
       i += (long long)gridDim.x * blockDim.x) {
    float mu = red[0] / nTot;
    float inv = rsqrtf(red[1] / nTot - mu * mu + EPSV);
    int c = (int)(i & 511);
    x[i] = (x[i] - mu) * inv * gam[c] + bet[c];
  }
}

// ---------------- pooling ----------------
__global__ void k_pool_add(const float* x, const long long* batch, float* pooled) {
  long long n = (long long)NN * 512;
  for (long long i = blockIdx.x * (long long)blockDim.x + threadIdx.x; i < n;
       i += (long long)gridDim.x * blockDim.x) {
    int node = (int)(i >> 9), c = (int)(i & 511);
    atomicAdd(&pooled[(size_t)batch[node] * 512 + c], x[i]);
  }
}
__global__ void k_pool_cnt(const long long* batch, float* cnt) {
  for (long long i = blockIdx.x * (long long)blockDim.x + threadIdx.x; i < NN;
       i += (long long)gridDim.x * blockDim.x)
    atomicAdd(&cnt[batch[i]], 1.0f);
}
__global__ void k_pool_div(float* pooled, const float* cnt) {
  int i = blockIdx.x * blockDim.x + threadIdx.x;
  if (i < NG * 512) pooled[i] /= fmaxf(cnt[i >> 9], 1.0f);
}

// ---------------- projection head (64 rows: tiny) ----------------
__global__ void k_proj1(const float* pooled, const float* w, const float* b, float* p1) {
  int i = blockIdx.x * blockDim.x + threadIdx.x;
  if (i >= NG * 128) return;
  int g = i >> 7, c = i & 127;
  float s = b[c];
  const float* pr = pooled + (size_t)g * 512;
  for (int k = 0; k < 512; ++k) s += pr[k] * w[k * 128 + c];
  p1[i] = s;
}
__global__ void k_bn64_gelu(float* p1, const float* gam, const float* bet) {
  int col = threadIdx.x; // 128 threads, 1 block
  float s = 0.f, q = 0.f;
  for (int g = 0; g < NG; ++g) { float v = p1[g * 128 + col]; s += v; q += v * v; }
  float mu = s / (float)NG, var = q / (float)NG - mu * mu;
  float inv = rsqrtf(var + EPSV);
  for (int g = 0; g < NG; ++g) {
    float v = (p1[g * 128 + col] - mu) * inv * gam[col] + bet[col];
    p1[g * 128 + col] = gelu_exact(v);
  }
}
__global__ void k_proj2(const float* p1, const float* w, const float* b, float* p2) {
  int i = blockIdx.x * blockDim.x + threadIdx.x;
  if (i >= NG * 128) return;
  int g = i >> 7, c = i & 127;
  float s = b[c] + p1[g * 128 + c];
  const float* pr = p1 + (size_t)g * 128;
  for (int k = 0; k < 128; ++k) s += pr[k] * w[k * 128 + c];
  p2[i] = s;
}
__global__ void k_rowln(float* p2, const float* gam, const float* bet) {
  int g = blockIdx.x, c = threadIdx.x; // 64 blocks x 128
  __shared__ float ss[128], sq[128];
  float v = p2[g * 128 + c];
  ss[c] = v; sq[c] = v * v; __syncthreads();
  for (int o = 64; o > 0; o >>= 1) {
    if (c < o) { ss[c] += ss[c + o]; sq[c] += sq[c + o]; }
    __syncthreads();
  }
  float mu = ss[0] / 128.0f, var = sq[0] / 128.0f - (ss[0] / 128.0f) * (ss[0] / 128.0f);
  p2[g * 128 + c] = (v - mu) * rsqrtf(var + EPSV) * gam[c] + bet[c];
}
__global__ void k_l2norm_out(const float* p2, float* out_x1) {
  int g = blockIdx.x, c = threadIdx.x; // 64 x 128
  __shared__ float sq[128];
  float v = p2[g * 128 + c];
  sq[c] = v * v; __syncthreads();
  for (int o = 64; o > 0; o >>= 1) {
    if (c < o) sq[c] += sq[c + o];
    __syncthreads();
  }
  float rn = 1.0f / fmaxf(sqrtf(sq[0]), 1e-12f);
  out_x1[g * 128 + c] = v * rn;
}
__global__ void k_logits(const float* x1, const float* w, const float* b, float* lg) {
  int i = blockIdx.x * blockDim.x + threadIdx.x;
  if (i >= NG * NCLS) return;
  int g = i / NCLS, c = i % NCLS;
  float s = b[c];
  const float* xr = x1 + (size_t)g * 128;
  for (int k = 0; k < 128; ++k) s += xr[k] * w[k * NCLS + c];
  lg[i] = s;
}
__global__ void k_log_softmax(const float* lg, float* out_c1) {
  int g = blockIdx.x, c = threadIdx.x; // 64 x 128 (100 active)
  __shared__ float sm[128];
  float v = (c < NCLS) ? lg[g * NCLS + c] : -3.4e38f;
  sm[c] = v; __syncthreads();
  for (int o = 64; o > 0; o >>= 1) {
    if (c < o) sm[c] = fmaxf(sm[c], sm[c + o]);
    __syncthreads();
  }
  float mx = sm[0]; __syncthreads();
  sm[c] = (c < NCLS) ? expf(v - mx) : 0.f; __syncthreads();
  for (int o = 64; o > 0; o >>= 1) {
    if (c < o) sm[c] += sm[c + o];
    __syncthreads();
  }
  float lse = logf(sm[0]);
  if (c < NCLS) out_c1[g * NCLS + c] = v - mx - lse;
}

// ---------------- host launch ----------------
static inline unsigned cap_grid(long long n, int t) {
  long long b = (n + t - 1) / t;
  if (b > 65535) b = 65535;
  if (b < 1) b = 1;
  return (unsigned)b;
}

extern "C" void kernel_launch(void* const* d_in, const int* in_sizes, int n_in,
                              void* d_out, int out_size, void* d_ws, size_t ws_size,
                              hipStream_t stream) {
  const float* x        = (const float*)d_in[0];
  const long long* ei   = (const long long*)d_in[1];
  const long long* bat  = (const long long*)d_in[2];
  const float* w_in     = (const float*)d_in[3];
  const float* g_in     = (const float*)d_in[5];
  const float* be_in    = (const float*)d_in[6];
  const float* w_h      = (const float*)d_in[7];
  const float* g_h      = (const float*)d_in[9];
  const float* be_h     = (const float*)d_in[10];
  const float* w_gat    = (const float*)d_in[11];
  const float* att_src  = (const float*)d_in[12];
  const float* att_dst  = (const float*)d_in[13];
  const float* b_gat    = (const float*)d_in[14];
  const float* w_out    = (const float*)d_in[15];
  const float* g_out    = (const float*)d_in[17];
  const float* be_out   = (const float*)d_in[18];
  const float* g_ln     = (const float*)d_in[19];
  const float* be_ln    = (const float*)d_in[20];
  const float* w_p1     = (const float*)d_in[21];
  const float* b_p1     = (const float*)d_in[22];
  const float* g_pbn    = (const float*)d_in[23];
  const float* be_pbn   = (const float*)d_in[24];
  const float* w_p2     = (const float*)d_in[25];
  const float* b_p2     = (const float*)d_in[26];
  const float* g_pln    = (const float*)d_in[27];
  const float* be_pln   = (const float*)d_in[28];
  const float* w_c      = (const float*)d_in[29];
  const float* b_c      = (const float*)d_in[30];
  (void)in_sizes; (void)n_in; (void)out_size; (void)ws_size;

  // ---- workspace carve ----
  char* base = (char*)d_ws;
  size_t off = 0;
  auto carve = [&](size_t bytes) -> char* {
    char* p = base + off;
    off = (off + bytes + 255) & ~(size_t)255;
    return p;
  };
  float* dis    = (float*)carve(sizeof(float) * NN);
  float* hbuf   = (float*)carve(sizeof(float) * (size_t)NN * 128);
  float* h2buf  = (float*)carve(sizeof(float) * (size_t)NN * 128);
  float* Gbuf   = (float*)carve(sizeof(float) * (size_t)NN * 128);
  float* O512   = (float*)carve(sizeof(float) * (size_t)NN * 512);
  float* coef   = (float*)carve(sizeof(float) * (size_t)NT * NH);
  float* asrc   = (float*)carve(sizeof(float) * (size_t)NN * NH);
  float* adst   = (float*)carve(sizeof(float) * (size_t)NN * NH);
  unsigned* mfl = (unsigned*)carve(sizeof(unsigned) * (size_t)NN * NH);
  float* denom  = (float*)carve(sizeof(float) * (size_t)NN * NH);
  float* was    = (float*)carve(sizeof(float) * 128 * NH);
  float* wad    = (float*)carve(sizeof(float) * 128 * NH);
  float* muv    = (float*)carve(sizeof(float) * 512);
  float* varv   = (float*)carve(sizeof(float) * 512);
  float* red2   = (float*)carve(sizeof(float) * 2);
  float* pooled = (float*)carve(sizeof(float) * NG * 512);
  float* cnt    = (float*)carve(sizeof(float) * NG);
  float* p1     = (float*)carve(sizeof(float) * NG * 128);
  float* p2     = (float*)carve(sizeof(float) * NG * 128);
  float* lg     = (float*)carve(sizeof(float) * NG * NCLS);
  __bf16* ABF   = (__bf16*)carve(sizeof(__bf16) * (size_t)NN * 128);
  __bf16* WBF   = (__bf16*)carve(sizeof(__bf16) * 128 * 512);

  float* out_x1 = (float*)d_out;
  float* out_c1 = (float*)d_out + NG * 128;

  const long long n128 = (long long)NN * 128;
  const long long n512 = (long long)NN * 512;
  dim3 B256(256);
  unsigned aggGrid = (unsigned)((NT + 1) / 2);

  // ---- degree normalization (self loops contribute exactly 1 -> init 1.0) ----
  k_fill_f32<<<cap_grid(NN, 256), B256, 0, stream>>>(dis, 1.0f, NN);
  k_deg_count<<<cap_grid(NE, 256), B256, 0, stream>>>(ei, dis);
  k_rsqrt_inplace<<<cap_grid(NN, 256), B256, 0, stream>>>(dis, NN);

  // ---- input GCN block: agg(3-dim) -> GEMM 3->128 -> BN -> GELU ----
  k_fill_f32<<<cap_grid(NN * 3, 256), B256, 0, stream>>>(Gbuf, 0.0f, (long long)NN * 3);
  k_agg3<<<cap_grid(NT, 256), B256, 0, stream>>>(ei, dis, x, Gbuf);
  k_lin3<<<cap_grid(n128, 256), B256, 0, stream>>>(Gbuf, w_in, hbuf);
  k_bn_stats<<<128, B256, 0, stream>>>(hbuf, 128, muv, varv);
  k_bn_gelu<<<cap_grid(n128, 256), B256, 0, stream>>>(hbuf, hbuf, (const float*)nullptr,
                                                      muv, varv, g_in, be_in, 128, n128);

  // ---- hidden GCN block (residual): agg -> WMMA GEMM 128->128 -> BN -> GELU -> +h ----
  k_fill_f32<<<cap_grid(n128, 256), B256, 0, stream>>>(Gbuf, 0.0f, n128);
  k_agg128<<<aggGrid, B256, 0, stream>>>(ei, dis, hbuf, Gbuf);
  k_to_bf16<<<cap_grid(n128, 256), B256, 0, stream>>>(ABF, Gbuf, n128);
  k_pack_b_wmma<<<cap_grid(128 * 128, 256), B256, 0, stream>>>(WBF, w_h, 128, 128, 0);
  {
    int tiles = (NN / 16) * (128 / 16);
    k_gemm128_bf16_wmma<<<(tiles + 7) / 8, B256, 0, stream>>>(ABF, WBF, Gbuf, NN, 128,
                                                              1.0f, 0);
  }
  k_bn_stats<<<128, B256, 0, stream>>>(Gbuf, 128, muv, varv);
  k_bn_gelu<<<cap_grid(n128, 256), B256, 0, stream>>>(Gbuf, hbuf, hbuf, muv, varv,
                                                      g_h, be_h, 128, n128);

  // ---- GAT (5 heads, mean) ----
  k_wa<<<(128 * NH + 255) / 256, B256, 0, stream>>>(w_gat, att_src, att_dst, was, wad);
  k_ah<<<cap_grid((long long)NN * NH, 256), B256, 0, stream>>>(hbuf, was, wad, asrc, adst);
  k_fill_u32<<<cap_grid((long long)NN * NH, 256), B256, 0, stream>>>(mfl, 0x007FFFFFu,
                                                                     (long long)NN * NH);
  k_fill_f32<<<cap_grid((long long)NN * NH, 256), B256, 0, stream>>>(denom, 0.0f,
                                                                     (long long)NN * NH);
  k_att_lrelu_max<<<cap_grid((long long)NT * NH, 256), B256, 0, stream>>>(ei, asrc, adst,
                                                                          coef, mfl);
  k_att_alpha<<<cap_grid((long long)NT * NH, 256), B256, 0, stream>>>(ei, coef, mfl, denom);
  k_fill_f32<<<cap_grid(n128, 256), B256, 0, stream>>>(h2buf, 0.0f, n128);
  for (int hd = 0; hd < NH; ++hd) {
    k_fill_f32<<<cap_grid(n128, 256), B256, 0, stream>>>(Gbuf, 0.0f, n128);
    k_agg_gat<<<aggGrid, B256, 0, stream>>>(ei, coef, denom, hbuf, Gbuf, hd);
    k_to_bf16<<<cap_grid(n128, 256), B256, 0, stream>>>(ABF, Gbuf, n128);
    k_pack_b_wmma<<<cap_grid(128 * 128, 256), B256, 0, stream>>>(WBF, w_gat, 128,
                                                                 NH * 128, hd * 128);
    int tiles = (NN / 16) * (128 / 16);
    // h2 += (1/5) * (S_h @ W_h)
    k_gemm128_bf16_wmma<<<(tiles + 7) / 8, B256, 0, stream>>>(ABF, WBF, h2buf, NN, 128,
                                                              0.2f, 1);
  }
  k_bias_add128<<<cap_grid(n128, 256), B256, 0, stream>>>(h2buf, b_gat, hbuf);

  // ---- output GCN block: agg -> WMMA GEMM 128->512 -> BN -> GELU ----
  k_fill_f32<<<cap_grid(n128, 256), B256, 0, stream>>>(Gbuf, 0.0f, n128);
  k_agg128<<<aggGrid, B256, 0, stream>>>(ei, dis, hbuf, Gbuf);
  k_to_bf16<<<cap_grid(n128, 256), B256, 0, stream>>>(ABF, Gbuf, n128);
  k_pack_b_wmma<<<cap_grid(128 * 512, 256), B256, 0, stream>>>(WBF, w_out, 512, 512, 0);
  {
    int tiles = (NN / 16) * (512 / 16);
    k_gemm128_bf16_wmma<<<(tiles + 7) / 8, B256, 0, stream>>>(ABF, WBF, O512, NN, 512,
                                                              1.0f, 0);
  }
  k_bn_stats<<<512, B256, 0, stream>>>(O512, 512, muv, varv);
  k_bn_gelu<<<cap_grid(n512, 256), B256, 0, stream>>>(O512, O512, (const float*)nullptr,
                                                      muv, varv, g_out, be_out, 512, n512);

  // ---- graph layer norm (global) ----
  k_fill_f32<<<1, dim3(2), 0, stream>>>(red2, 0.0f, 2);
  k_red2<<<cap_grid(n512, 256), B256, 0, stream>>>(O512, n512, red2);
  k_gln_apply<<<cap_grid(n512, 256), B256, 0, stream>>>(O512, red2, g_ln, be_ln);

  // ---- global mean pool ----
  k_fill_f32<<<cap_grid(NG * 512, 256), B256, 0, stream>>>(pooled, 0.0f, NG * 512);
  k_fill_f32<<<1, dim3(NG), 0, stream>>>(cnt, 0.0f, NG);
  k_pool_add<<<cap_grid(n512, 256), B256, 0, stream>>>(O512, bat, pooled);
  k_pool_cnt<<<cap_grid(NN, 256), B256, 0, stream>>>(bat, cnt);
  k_pool_div<<<(NG * 512 + 255) / 256, B256, 0, stream>>>(pooled, cnt);

  // ---- projection head + classifier (tiny) ----
  k_proj1<<<(NG * 128 + 255) / 256, B256, 0, stream>>>(pooled, w_p1, b_p1, p1);
  k_bn64_gelu<<<1, dim3(128), 0, stream>>>(p1, g_pbn, be_pbn);
  k_proj2<<<(NG * 128 + 255) / 256, B256, 0, stream>>>(p1, w_p2, b_p2, p2);
  k_rowln<<<NG, dim3(128), 0, stream>>>(p2, g_pln, be_pln);
  k_l2norm_out<<<NG, dim3(128), 0, stream>>>(p2, out_x1);
  k_logits<<<(NG * NCLS + 255) / 256, B256, 0, stream>>>(out_x1, w_c, b_c, lg);
  k_log_softmax<<<NG, dim3(128), 0, stream>>>(lg, out_c1);
}
